// SpatialTransformer_5308579578324
// MI455X (gfx1250) — compile-verified, matrix operand order checked by reading, not run on hardware
//
#include <hip/hip_runtime.h>
#include <hip/hip_bf16.h>

// ---------------------------------------------------------------------------
// Video Swin block x2 for gfx1250 (MI455X).  All GEMMs via
// v_wmma_f32_16x16x32_f16 (wave32, 16x16 tiles, K-step 32).  GEMM A-tiles are
// staged into LDS with the CDNA5 async copy path
// (global_load_async_to_lds_b128 + s_wait_asynccnt), double-buffered.
// ---------------------------------------------------------------------------

typedef _Float16 v8h  __attribute__((ext_vector_type(8)));
typedef _Float16 v16h __attribute__((ext_vector_type(16)));
typedef float    v8f  __attribute__((ext_vector_type(8)));

union HF16 { v16h v; v8h h[2]; };

__device__ __forceinline__ v8f wmma16x16x32(v16h a, v16h b, v8f c) {
  return __builtin_amdgcn_wmma_f32_16x16x32_f16(false, a, false, b, (short)0, c,
                                                false, false);
}

static constexpr int   Bn = 2, Cc = 96, Dd = 32, Hh = 64, Ww = 64;
static constexpr int   NH = 3, HD = 32, Nw = 64;           // window tokens
static constexpr long  Ttok = (long)Bn * Dd * Hh * Ww;     // 262144
static constexpr float SCALE = 0.17677669529663687f;       // 32^-0.5

// issue one 16-byte async copy from global to LDS (ASYNCcnt-tracked DMA)
__device__ __forceinline__ void async_copy_b128(unsigned lds_off,
                                                const void* gaddr) {
  asm volatile("global_load_async_to_lds_b128 %0, %1, off"
               :: "v"(lds_off), "v"(gaddr) : "memory");
}

// -------------------------------- permutes ---------------------------------
__global__ void k_permute_in(const float* __restrict__ x, float* __restrict__ o) {
  size_t i = (size_t)blockIdx.x * blockDim.x + threadIdx.x; // (b,c,d,h,w) order
  int wv = (int)(i % 64); size_t r = i / 64;
  int hh = (int)(r % 64); r /= 64;
  int dd = (int)(r % 32); r /= 32;
  int cc = (int)(r % 96); int bb = (int)(r / 96);
  size_t t = ((size_t)((bb * 32 + dd) * 64 + hh) * 64 + wv);
  o[t * 96 + cc] = x[i];
}

__global__ void k_permute_out(const float* __restrict__ x, float* __restrict__ o) {
  size_t i = (size_t)blockIdx.x * blockDim.x + threadIdx.x; // (b,c,d,h,w) order
  int wv = (int)(i % 64); size_t r = i / 64;
  int hh = (int)(r % 64); r /= 64;
  int dd = (int)(r % 32); r /= 32;
  int cc = (int)(r % 96); int bb = (int)(r / 96);
  size_t t = ((size_t)((bb * 32 + dd) * 64 + hh) * 64 + wv);
  o[i] = x[t * 96 + cc];
}

__global__ void k_cvt_f16(const float* __restrict__ in, _Float16* __restrict__ out,
                          int n) {
  int i = blockIdx.x * blockDim.x + threadIdx.x;
  if (i < n) out[i] = (_Float16)in[i];
}

// ------------------------ fused LayerNorm (+shift+window) ------------------
__global__ void k_layernorm(const float* __restrict__ src,
                            const float* __restrict__ g,
                            const float* __restrict__ bta,
                            _Float16* __restrict__ dst,
                            int windowed, int shifted) {
  int t = blockIdx.x * blockDim.x + threadIdx.x;
  int s;
  if (windowed) {
    int w = t >> 6, n = t & 63;
    int b = w >> 11, wib = w & 2047;
    int bd = wib >> 8, bh = (wib >> 4) & 15, bw = wib & 15;
    int dz = n >> 4, hy = (n >> 2) & 3, wx = n & 3;
    int d = bd * 4 + dz, hh = bh * 4 + hy, ww = bw * 4 + wx;
    if (shifted) { d = (d + 2) & 31; hh = (hh + 2) & 63; ww = (ww + 2) & 63; }
    s = ((b * 32 + d) * 64 + hh) * 64 + ww;
  } else {
    s = t;
  }
  const float* xr = src + (size_t)s * 96;
  float mu = 0.f;
  for (int c = 0; c < 96; ++c) mu += xr[c];
  mu *= (1.f / 96.f);
  float var = 0.f;
  for (int c = 0; c < 96; ++c) { float d0 = xr[c] - mu; var += d0 * d0; }
  var *= (1.f / 96.f);
  float rs = rsqrtf(var + 1e-5f);
  _Float16* o = dst + (size_t)t * 96;
  for (int c = 0; c < 96; ++c)
    o[c] = (_Float16)((xr[c] - mu) * rs * g[c] + bta[c]);
}

// ------------------------------- WMMA GEMM ---------------------------------
// C[M,N] = A[M,K] * Bt[N,K]^T + bias.  A k-slices (64x32 f16) staged into LDS
// by the async copy engine, double-buffered; WMMA consumes from LDS.
// EPI: 0=f16 out, 1=f16 out + exact GELU, 2=f32 out + residual (winmap:
// scatter through window-reverse + unshift).
template <int EPI>
__global__ void k_gemm_wmma(const _Float16* __restrict__ A,
                            const _Float16* __restrict__ Bt,
                            const float* __restrict__ bias,
                            void* __restrict__ outp,
                            const float* __restrict__ res,
                            int N, int K, int winmap, int shifted) {
  __shared__ __align__(16) _Float16 Ash[2][64][32]; // 2 x 4KB k-slices
  const int tid = threadIdx.x;
  const int lane = tid & 31, wv = tid >> 5;
  const int grp = lane >> 4, l16 = lane & 15;
  const int Mbase = blockIdx.x * 64;
  const int ncol = blockIdx.y * 32 + (wv >> 2) * 16 + l16; // B col / out col
  const _Float16* Br = Bt + (size_t)ncol * K;
  __builtin_prefetch(Br, 0, 3);

  // async staging assignment: 256 thr x 16B = one 64x32 f16 slice
  const int cr = tid >> 2;            // row 0..63
  const int cco = (tid & 3) * 8;      // half-offset 0,8,16,24
  const _Float16* Ag = A + (size_t)(Mbase + cr) * K + cco;
  const unsigned lds0 = (unsigned)(size_t)&Ash[0][cr][cco];
  const unsigned lds1 = (unsigned)(size_t)&Ash[1][cr][cco];

  async_copy_b128(lds0, Ag);          // prologue: slice for kk=0 -> buf 0

  v8f acc = {0.f, 0.f, 0.f, 0.f, 0.f, 0.f, 0.f, 0.f};
  int buf = 0;
  for (int kk = 0; kk < K; kk += 32) {
    if (kk + 32 < K) {
      async_copy_b128(buf ? lds0 : lds1, Ag + kk + 32); // prefetch next slice
      asm volatile("s_wait_asynccnt 0x1" ::: "memory"); // oldest (cur) landed
    } else {
      asm volatile("s_wait_asynccnt 0x0" ::: "memory");
    }
    __syncthreads();                  // slice visible to all waves
    HF16 a, b;
    const _Float16* arow = &Ash[buf][(wv & 3) * 16 + l16][grp * 8];
    a.h[0] = *(const v8h*)arow;
    a.h[1] = *(const v8h*)(arow + 16);
    b.h[0] = *(const v8h*)(Br + kk + grp * 8);
    b.h[1] = *(const v8h*)(Br + kk + 16 + grp * 8);
    acc = wmma16x16x32(a.v, b.v, acc);
    __syncthreads();                  // protect slice vs next async overwrite
    buf ^= 1;
  }

  const float bn = bias[ncol];
  const int mbase = Mbase + (wv & 3) * 16 + grp * 8; // D-frag rows
  if (EPI <= 1) {
    _Float16* out = (_Float16*)outp;
#pragma unroll
    for (int j = 0; j < 8; ++j) {
      float v = acc[j] + bn;
      if (EPI == 1) v = 0.5f * v * (1.f + erff(v * 0.70710678118f));
      out[(size_t)(mbase + j) * N + ncol] = (_Float16)v;
    }
  } else {
    float* out = (float*)outp;
#pragma unroll
    for (int j = 0; j < 8; ++j) {
      int t = mbase + j;
      int dstT = t;
      if (winmap) {
        int w = t >> 6, n = t & 63;
        int b = w >> 11, wib = w & 2047;
        int bd = wib >> 8, bh = (wib >> 4) & 15, bw = wib & 15;
        int dz = n >> 4, hy = (n >> 2) & 3, wx = n & 3;
        int d = bd * 4 + dz, hh = bh * 4 + hy, ww = bw * 4 + wx;
        if (shifted) { d = (d + 2) & 31; hh = (hh + 2) & 63; ww = (ww + 2) & 63; }
        dstT = ((b * 32 + d) * 64 + hh) * 64 + ww;
      }
      size_t o = (size_t)dstT * 96 + ncol;  // N == 96 for EPI 2
      out[o] = res[o] + acc[j] + bn;
    }
  }
}

// ------------------------------- attention ---------------------------------
__device__ __forceinline__ int zone3(int g, int dim) {
  return g >= dim - 2 ? 2 : (g >= dim - 4 ? 1 : 0);
}

// One block (128 thr = 4 waves) per (window, head).  Each wave owns a
// 16-row strip of the 64x64 score matrix: Q*K^T via WMMA, softmax with
// rel-pos bias (+shift mask) via wave32 shuffles, P and V^T staged in LDS,
// then P@V via WMMA.
__global__ void k_attention(const _Float16* __restrict__ qkv,
                            const float* __restrict__ rpb,
                            _Float16* __restrict__ out, int shifted) {
  const int w = blockIdx.x, h = blockIdx.y;
  const int lane = threadIdx.x & 31, wv = threadIdx.x >> 5;
  const int grp = lane >> 4, l16 = lane & 15;

  __shared__ __align__(16) _Float16 Pbuf[4][16][72]; // 16B-aligned rows (144B)
  __shared__ __align__(16) _Float16 VT[32][72];      // V transposed [d][n]

  // stage V^T cooperatively
  for (int idx = threadIdx.x; idx < Nw * HD; idx += 128) {
    int n = idx >> 5, d = idx & 31;
    VT[d][n] = qkv[((size_t)(w * 64 + n)) * 288 + 192 + h * 32 + d];
  }
  __syncthreads();

  // ---- S = (Q * scale) K^T : one 16x64 strip per wave, K-dim = HD = 32 ----
  HF16 aq;
  const _Float16* qrow = qkv + ((size_t)(w * 64 + wv * 16 + l16)) * 288 + h * 32;
  aq.h[0] = *(const v8h*)(qrow + grp * 8);
  aq.h[1] = *(const v8h*)(qrow + 16 + grp * 8);
  v8f s[4];
#pragma unroll
  for (int nt = 0; nt < 4; ++nt) {
    HF16 bk;
    const _Float16* krow =
        qkv + ((size_t)(w * 64 + nt * 16 + l16)) * 288 + 96 + h * 32;
    bk.h[0] = *(const v8h*)(krow + grp * 8);
    bk.h[1] = *(const v8h*)(krow + 16 + grp * 8);
    v8f z = {0.f, 0.f, 0.f, 0.f, 0.f, 0.f, 0.f, 0.f};
    s[nt] = wmma16x16x32(aq.v, bk.v, z);
  }

  // ---- bias + mask + row softmax (rows split across 16-lane halves) ----
  const int wib = w & 2047;
  const int bd = wib >> 8, bh = (wib >> 4) & 15, bw = wib & 15;
#pragma unroll
  for (int j = 0; j < 8; ++j) {
    const int m = wv * 16 + j + grp * 8;
    const int mz = m >> 4, my = (m >> 2) & 3, mx = m & 3;
    int zm = 0;
    if (shifted)
      zm = zone3(bd * 4 + mz, 32) * 9 + zone3(bh * 4 + my, 64) * 3 +
           zone3(bw * 4 + mx, 64);
    float vals[4];
    float rmax = -3.0e38f;
#pragma unroll
    for (int nt = 0; nt < 4; ++nt) {
      const int n = nt * 16 + l16;
      const int nz = n >> 4, ny = (n >> 2) & 3, nx = n & 3;
      const int ridx = (mz - nz + 3) * 49 + (my - ny + 3) * 7 + (mx - nx + 3);
      float v = s[nt][j] * SCALE + rpb[ridx * 3 + h];
      if (shifted) {
        int zn = zone3(bd * 4 + nz, 32) * 9 + zone3(bh * 4 + ny, 64) * 3 +
                 zone3(bw * 4 + nx, 64);
        if (zn != zm) v -= 100.f;
      }
      vals[nt] = v;
      rmax = fmaxf(rmax, v);
    }
#pragma unroll
    for (int off = 1; off < 16; off <<= 1)
      rmax = fmaxf(rmax, __shfl_xor(rmax, off, 32));
    float rsum = 0.f;
#pragma unroll
    for (int nt = 0; nt < 4; ++nt) {
      vals[nt] = __expf(vals[nt] - rmax);
      rsum += vals[nt];
    }
#pragma unroll
    for (int off = 1; off < 16; off <<= 1) rsum += __shfl_xor(rsum, off, 32);
    const float inv = 1.f / rsum;
#pragma unroll
    for (int nt = 0; nt < 4; ++nt)
      Pbuf[wv][j + grp * 8][nt * 16 + l16] = (_Float16)(vals[nt] * inv);
  }
  __syncthreads();

  // ---- O = P @ V : 16x32 strip, two K-steps of 32 over the 64 keys ----
  v8f o0 = {0.f, 0.f, 0.f, 0.f, 0.f, 0.f, 0.f, 0.f};
  v8f o1 = {0.f, 0.f, 0.f, 0.f, 0.f, 0.f, 0.f, 0.f};
#pragma unroll
  for (int ks = 0; ks < 2; ++ks) {
    HF16 ap, bv;
    const _Float16* prow = &Pbuf[wv][l16][ks * 32 + grp * 8];
    ap.h[0] = *(const v8h*)prow;
    ap.h[1] = *(const v8h*)(prow + 16);
    const _Float16* v0 = &VT[l16][ks * 32 + grp * 8];
    bv.h[0] = *(const v8h*)v0;
    bv.h[1] = *(const v8h*)(v0 + 16);
    o0 = wmma16x16x32(ap.v, bv.v, o0);
    const _Float16* v1 = &VT[16 + l16][ks * 32 + grp * 8];
    bv.h[0] = *(const v8h*)v1;
    bv.h[1] = *(const v8h*)(v1 + 16);
    o1 = wmma16x16x32(ap.v, bv.v, o1);
  }
#pragma unroll
  for (int j = 0; j < 8; ++j) {
    const int m = wv * 16 + j + grp * 8;
    const size_t rowo = ((size_t)(w * 64 + m)) * 96 + h * 32;
    out[rowo + l16]      = (_Float16)o0[j];
    out[rowo + 16 + l16] = (_Float16)o1[j];
  }
}

// -------------------------------- launcher ---------------------------------
extern "C" void kernel_launch(void* const* d_in, const int* in_sizes, int n_in,
                              void* d_out, int out_size, void* d_ws,
                              size_t ws_size, hipStream_t stream) {
  (void)in_sizes; (void)n_in; (void)out_size; (void)ws_size;
  const float* x       = (const float*)d_in[0];
  const float* n1g     = (const float*)d_in[1];
  const float* n1b     = (const float*)d_in[2];
  const float* qkv_w   = (const float*)d_in[3];
  const float* qkv_b   = (const float*)d_in[4];
  const float* rpb     = (const float*)d_in[5];
  const float* proj_w  = (const float*)d_in[6];
  const float* proj_b  = (const float*)d_in[7];
  const float* n2g     = (const float*)d_in[8];
  const float* n2b     = (const float*)d_in[9];
  const float* fc1_w   = (const float*)d_in[10];
  const float* fc1_b   = (const float*)d_in[11];
  const float* fc2_w   = (const float*)d_in[12];
  const float* fc2_b   = (const float*)d_in[13];
  float* outp = (float*)d_out;

  char* ws = (char*)d_ws;
  const size_t T = (size_t)Ttok;
  size_t off = 0;
  float*    XA  = (float*)(ws + off); off += T * 96 * 4;
  float*    XB  = (float*)(ws + off); off += T * 96 * 4;
  float*    XC  = (float*)(ws + off); off += T * 96 * 4;
  _Float16* XW  = (_Float16*)(ws + off); off += T * 96 * 2;   // LN out / act
  _Float16* QKV = (_Float16*)(ws + off); off += T * 384 * 2;  // qkv, later hid
  _Float16* AO  = (_Float16*)(ws + off); off += T * 96 * 2;   // attn out
  _Float16* Wq  = (_Float16*)(ws + off); off += 2 * 288 * 96 * 2;
  _Float16* Wp  = (_Float16*)(ws + off); off += 2 * 96 * 96 * 2;
  _Float16* W1  = (_Float16*)(ws + off); off += 2 * 384 * 96 * 2;
  _Float16* W2  = (_Float16*)(ws + off); off += 2 * 96 * 384 * 2;

  // weights -> f16 (once per call; deterministic)
  k_cvt_f16<<<(2 * 288 * 96 + 255) / 256, 256, 0, stream>>>(qkv_w, Wq, 2 * 288 * 96);
  k_cvt_f16<<<(2 * 96 * 96 + 255) / 256, 256, 0, stream>>>(proj_w, Wp, 2 * 96 * 96);
  k_cvt_f16<<<(2 * 384 * 96 + 255) / 256, 256, 0, stream>>>(fc1_w, W1, 2 * 384 * 96);
  k_cvt_f16<<<(2 * 96 * 384 + 255) / 256, 256, 0, stream>>>(fc2_w, W2, 2 * 96 * 384);

  const size_t nelem = T * 96;
  k_permute_in<<<(unsigned)(nelem / 256), 256, 0, stream>>>(x, XA);

  for (int i = 0; i < 2; ++i) {
    const int sh = i; // block 1 shifted
    float* Xin  = i ? XC : XA;
    float* Xatt = i ? XA : XB;
    float* Xmlp = i ? XB : XC;

    // LN1 + (shift) + window partition -> f16
    k_layernorm<<<(unsigned)(T / 256), 256, 0, stream>>>(
        Xin, n1g + i * 96, n1b + i * 96, XW, 1, sh);
    // QKV: [T,96] x [288,96]^T
    k_gemm_wmma<0><<<dim3((unsigned)(T / 64), 288 / 32), 256, 0, stream>>>(
        XW, Wq + (size_t)i * 288 * 96, qkv_b + i * 288, QKV, nullptr, 288, 96, 0, 0);
    // attention per (window, head)
    k_attention<<<dim3(4096, 3), 128, 0, stream>>>(
        QKV, rpb + (size_t)i * 343 * 3, AO, sh);
    // proj + residual, scattered through window-reverse (+unshift)
    k_gemm_wmma<2><<<dim3((unsigned)(T / 64), 96 / 32), 256, 0, stream>>>(
        AO, Wp + (size_t)i * 96 * 96, proj_b + i * 96, Xatt, Xin, 96, 96, 1, sh);
    // LN2 (plain order)
    k_layernorm<<<(unsigned)(T / 256), 256, 0, stream>>>(
        Xatt, n2g + i * 96, n2b + i * 96, XW, 0, 0);
    // FC1 + exact GELU -> hid (reuses QKV buffer)
    k_gemm_wmma<1><<<dim3((unsigned)(T / 64), 384 / 32), 256, 0, stream>>>(
        XW, W1 + (size_t)i * 384 * 96, fc1_b + i * 384, QKV, nullptr, 384, 96, 0, 0);
    // FC2 + residual
    k_gemm_wmma<2><<<dim3((unsigned)(T / 64), 96 / 32), 256, 0, stream>>>(
        QKV, W2 + (size_t)i * 96 * 384, fc2_b + i * 96, Xmlp, Xatt, 96, 384, 0, 0);
  }

  k_permute_out<<<(unsigned)(nelem / 256), 256, 0, stream>>>(XB, outp);
}
